// ActionSelector_83288005804352
// MI455X (gfx1250) — matrix operand order repaired and checked by reading, not compiled, not run on hardware
//
#include <hip/hip_runtime.h>
#include <hip/hip_bf16.h>

// ---------------- problem constants ----------------
#define N_GRAPHS        12500
#define NODES_PER_GRAPH 40
#define N_NODES         500000
#define D_NODE          128
#define D_CTX           64
#define H1DIM           256
#define H2DIM           128

#define KT1 10                        // 320 / 32
#define NT1 16                        // 256 / 16
#define KT2 8                         // 256 / 32
#define NT2 8                         // 128 / 16

#define W1_FRAG_HALVES (NT1 * KT1 * 32 * 16)   // 81920 halves = 160 KB
#define W2_FRAG_HALVES (NT2 * KT2 * 32 * 16)   // 32768 halves =  64 KB
#define N_CHUNKS  (N_NODES / 32)               // 15625 chunks of 32 nodes
#define H1_PAIR_HALVES (32 * H1DIM)            // 8192 halves = 16 KB per wave-pair
#define GRID_BLOCKS 512

typedef __attribute__((ext_vector_type(16))) _Float16 v16h;
typedef __attribute__((ext_vector_type(8)))  float    v8f;

union U32h { uint4 q[2]; v16h h; };
union U8f  { float4 q[2]; float f[8]; };

// ---------------------------------------------------------------------------
// Prep: convert W1 [320x256] and W2 [256x128] f32 -> f16, pre-swizzled into
// the per-lane WMMA B-fragment layout (lane L: column N = L%16, h = L/16,
// half element e -> K = 8h + e for e < 8, K = 16 + 8h + (e - 8) for e >= 8).
// Fragment flat index: (((tn*KT + kt)*32) + lane)*16 + e
// ---------------------------------------------------------------------------
__global__ void prep_weights_kernel(const float* __restrict__ W1,
                                    const float* __restrict__ W2,
                                    _Float16* __restrict__ frag) {
    int idx = blockIdx.x * blockDim.x + threadIdx.x;
    if (idx >= W1_FRAG_HALVES + W2_FRAG_HALVES) return;
    if (idx < W1_FRAG_HALVES) {
        int t = idx >> 9;
        int r = idx & 511;
        int lane = r >> 4, e = r & 15;
        int tn = t / KT1, kt = t % KT1;
        int K = kt * 32 + ((e < 8) ? (lane >> 4) * 8 + e : 16 + (lane >> 4) * 8 + (e - 8));
        int N = tn * 16 + (lane & 15);
        frag[idx] = (_Float16)W1[K * H1DIM + N];
    } else {
        int j = idx - W1_FRAG_HALVES;
        int t = j >> 9;
        int r = j & 511;
        int lane = r >> 4, e = r & 15;
        int tn = t / KT2, kt = t % KT2;
        int K = kt * 32 + ((e < 8) ? (lane >> 4) * 8 + e : 16 + (lane >> 4) * 8 + (e - 8));
        int N = tn * 16 + (lane & 15);
        frag[idx] = (_Float16)W2[K * H2DIM + N];
    }
}

// ---------------------------------------------------------------------------
// Fused MLP. A pair of waves cooperates on a 32-node chunk (two 16-row
// M-tiles): both hold the A fragments for both tiles; each wave computes half
// of the N-tiles, so every B-fragment LDS load feeds TWO WMMAs.
// ---------------------------------------------------------------------------
__launch_bounds__(256)
__global__ void mlp_fused_kernel(const float* __restrict__ nf,
                                 const int*   __restrict__ prev_act,
                                 const float* __restrict__ ctx,
                                 const int*   __restrict__ n2g,
                                 const int*   __restrict__ goff,
                                 const _Float16* __restrict__ wfrag,
                                 const float* __restrict__ b1,
                                 const float* __restrict__ b2,
                                 const float* __restrict__ W3,
                                 const float* __restrict__ b3,
                                 float* __restrict__ out) {
    extern __shared__ char smem[];
    _Float16* lw1  = (_Float16*)smem;                       // 160 KB
    _Float16* lw2  = lw1 + W1_FRAG_HALVES;                  //  64 KB
    _Float16* lh1  = lw2 + W2_FRAG_HALVES;                  //  64 KB (4 pairs x 16 KB)
    float*    lpart = (float*)(lh1 + 4 * H1_PAIR_HALVES);   //   1 KB (4 pairs x 64 f32)

    // ---- cooperative weight copy into LDS (229376 B = 14336 uint4) ----
    {
        const uint4* src = (const uint4*)wfrag;
        uint4* dst = (uint4*)smem;
        const int nvec = (W1_FRAG_HALVES + W2_FRAG_HALVES) / 8;
        for (int i = threadIdx.x; i < nvec; i += blockDim.x) dst[i] = src[i];
    }
    __syncthreads();

    const int lane = threadIdx.x & 31;
    const int wave = threadIdx.x >> 5;
    const int pair = wave >> 1;     // 4 pairs per block
    const int half = wave & 1;      // which half of the N-tiles this wave owns
    const int col  = lane & 15;
    const int hi   = lane >> 4;
    const int c    = hi * 8;

    _Float16* h1p   = lh1 + pair * H1_PAIR_HALVES;          // 32 x 256 f16
    float*    partw = lpart + (pair * 2 + half) * 32;       // this wave's partials
    float*    partp = lpart + pair * 64;                    // pair base
    const uint4* w1q = (const uint4*)lw1;
    const uint4* w2q = (const uint4*)lw2;
    const float b3v = b3[0];

    const int stride = gridDim.x * 4;                       // chunks per grid step
    const int iters  = (N_CHUNKS + stride - 1) / stride;    // uniform for all waves
    int chunk = blockIdx.x * 4 + pair;

    for (int it = 0; it < iters; ++it, chunk += stride) {
        const bool valid = (chunk < N_CHUNKS);

        if (valid) {
            // ---- build layer-1 A fragments for BOTH 16-row tiles ----
            v16h a1[2][KT1];
#pragma unroll
            for (int t = 0; t < 2; ++t) {
                const int node  = chunk * 32 + t * 16 + col;
                const int g     = n2g[node];
                const int pnode = goff[g] + prev_act[g];
                const float* rowNF = nf  + (size_t)node  * D_NODE;
                const float* rowPV = nf  + (size_t)pnode * D_NODE;
                const float* rowCX = ctx + (size_t)g     * D_CTX;
#pragma unroll
                for (int kt = 0; kt < KT1; ++kt) {
                    const float* base = (kt < 4) ? (rowNF + kt * 32)
                                      : (kt < 8) ? (rowPV + (kt - 4) * 32)
                                                 : (rowCX + (kt - 8) * 32);
                    U8f s1, s2;
                    s1.q[0] = ((const float4*)(base + c))[0];
                    s1.q[1] = ((const float4*)(base + c))[1];
                    s2.q[0] = ((const float4*)(base + c + 16))[0];
                    s2.q[1] = ((const float4*)(base + c + 16))[1];
                    v16h a;
#pragma unroll
                    for (int e = 0; e < 8; ++e) {
                        a[e]     = (_Float16)s1.f[e];
                        a[8 + e] = (_Float16)s2.f[e];
                    }
                    a1[t][kt] = a;
                }
            }

            // ---- prefetch next chunk's streaming node-feature rows ----
            {
                const int cn = chunk + stride;
                if (cn < N_CHUNKS) {
#pragma unroll
                    for (int t = 0; t < 2; ++t) {
                        const float* p =
                            nf + (size_t)(cn * 32 + t * 16 + col) * D_NODE + hi * 64;
                        __builtin_prefetch(p, 0, 1);
                        __builtin_prefetch(p + 32, 0, 1);
                    }
                }
            }

            // ---- layer 1: this wave's 8 n-tiles, both M-tiles per B load ----
            for (int tn8 = 0; tn8 < 8; ++tn8) {
                const int tn = half * 8 + tn8;
                v8f acc0 = {}, acc1 = {};
#pragma unroll
                for (int kt = 0; kt < KT1; ++kt) {
                    U32h b;
                    const int fi = ((tn * KT1 + kt) * 32 + lane) * 2;
                    b.q[0] = w1q[fi];
                    b.q[1] = w1q[fi + 1];
                    acc0 = __builtin_amdgcn_wmma_f32_16x16x32_f16(
                        false, a1[0][kt], false, b.h, (short)0, acc0, false, false);
                    acc1 = __builtin_amdgcn_wmma_f32_16x16x32_f16(
                        false, a1[1][kt], false, b.h, (short)0, acc1, false, false);
                }
                const float bb = b1[tn * 16 + col];
#pragma unroll
                for (int v = 0; v < 8; ++v) {
                    float h0 = acc0[v] + bb; h0 = h0 > 0.f ? h0 : 0.f;
                    float h1v = acc1[v] + bb; h1v = h1v > 0.f ? h1v : 0.f;
                    h1p[(v + 8 * hi) * H1DIM + tn * 16 + col]        = (_Float16)h0;
                    h1p[(16 + v + 8 * hi) * H1DIM + tn * 16 + col]   = (_Float16)h1v;
                }
            }
        }
        __syncthreads();   // h1 halves from both waves of the pair visible

        if (valid) {
            // ---- layer-2 A fragments for both tiles from staged h1 ----
            v16h a2[2][KT2];
#pragma unroll
            for (int t = 0; t < 2; ++t) {
#pragma unroll
                for (int kt = 0; kt < KT2; ++kt) {
                    U32h a;
                    const _Float16* p = h1p + (t * 16 + col) * H1DIM + kt * 32 + c;
                    a.q[0] = *(const uint4*)p;
                    a.q[1] = *(const uint4*)(p + 16);
                    a2[t][kt] = a.h;
                }
            }

            // ---- layer 2 (this wave's 4 n2-tiles) + fused layer 3 partials ----
            float sum0[8], sum1[8];
#pragma unroll
            for (int v = 0; v < 8; ++v) { sum0[v] = 0.f; sum1[v] = 0.f; }

            for (int tj = 0; tj < 4; ++tj) {
                const int tn2 = half * 4 + tj;
                v8f acc0 = {}, acc1 = {};
#pragma unroll
                for (int kt = 0; kt < KT2; ++kt) {
                    U32h b;
                    const int fi = ((tn2 * KT2 + kt) * 32 + lane) * 2;
                    b.q[0] = w2q[fi];
                    b.q[1] = w2q[fi + 1];
                    acc0 = __builtin_amdgcn_wmma_f32_16x16x32_f16(
                        false, a2[0][kt], false, b.h, (short)0, acc0, false, false);
                    acc1 = __builtin_amdgcn_wmma_f32_16x16x32_f16(
                        false, a2[1][kt], false, b.h, (short)0, acc1, false, false);
                }
                const float bb  = b2[tn2 * 16 + col];
                const float w3v = W3[tn2 * 16 + col];
#pragma unroll
                for (int v = 0; v < 8; ++v) {
                    float h0 = acc0[v] + bb; h0 = h0 > 0.f ? h0 : 0.f;
                    float h1v = acc1[v] + bb; h1v = h1v > 0.f ? h1v : 0.f;
                    sum0[v] += h0 * w3v;
                    sum1[v] += h1v * w3v;
                }
            }

            // ---- reduce across the 16 columns of each half-wave ----
#pragma unroll
            for (int v = 0; v < 8; ++v) {
                float s0 = sum0[v], s1 = sum1[v];
                s0 += __shfl_xor(s0, 1, 32);  s1 += __shfl_xor(s1, 1, 32);
                s0 += __shfl_xor(s0, 2, 32);  s1 += __shfl_xor(s1, 2, 32);
                s0 += __shfl_xor(s0, 4, 32);  s1 += __shfl_xor(s1, 4, 32);
                s0 += __shfl_xor(s0, 8, 32);  s1 += __shfl_xor(s1, 8, 32);
                if (col == 0) {
                    partw[v + 8 * hi]      = s0;   // tile 0 rows
                    partw[16 + v + 8 * hi] = s1;   // tile 1 rows
                }
            }
        }
        __syncthreads();   // partials from both waves of the pair visible

        if (valid && half == 0) {
            // combine the two waves' n2-halves and emit 32 logits
            const float vsum = partp[lane] + partp[32 + lane] + b3v;
            out[chunk * 32 + lane] = vsum;
        }
    }
}

// ---------------------------------------------------------------------------
extern "C" void kernel_launch(void* const* d_in, const int* in_sizes, int n_in,
                              void* d_out, int out_size, void* d_ws, size_t ws_size,
                              hipStream_t stream) {
    const float* nf    = (const float*)d_in[0];   // node_features [500000,128]
    const int*   prev  = (const int*)  d_in[1];   // prev_action_per_graph [12500]
    const float* ctx   = (const float*)d_in[2];   // context [12500,64]
    const int*   n2g   = (const int*)  d_in[3];   // node_to_graphid [500000]
    const int*   goff  = (const int*)  d_in[4];   // graph_offsets [12500]
    const float* W1    = (const float*)d_in[5];   // [320,256]
    const float* b1    = (const float*)d_in[6];   // [256]
    const float* W2    = (const float*)d_in[7];   // [256,128]
    const float* b2    = (const float*)d_in[8];   // [128]
    const float* W3    = (const float*)d_in[9];   // [128,1]
    const float* b3    = (const float*)d_in[10];  // [1]
    float* out = (float*)d_out;

    _Float16* frag = (_Float16*)d_ws;             // 229376 bytes of scratch

    const int total_halves = W1_FRAG_HALVES + W2_FRAG_HALVES;   // 114688
    prep_weights_kernel<<<(total_halves + 255) / 256, 256, 0, stream>>>(W1, W2, frag);

    // LDS: 224 KB weights + 64 KB h1 staging + 1 KB partials = 295936 B
    const size_t lds_bytes = (size_t)(W1_FRAG_HALVES + W2_FRAG_HALVES) * 2
                           + (size_t)4 * H1_PAIR_HALVES * 2
                           + (size_t)4 * 64 * sizeof(float);
    hipFuncSetAttribute((const void*)mlp_fused_kernel,
                        hipFuncAttributeMaxDynamicSharedMemorySize, (int)lds_bytes);
    mlp_fused_kernel<<<GRID_BLOCKS, 256, lds_bytes, stream>>>(
        nf, prev, ctx, n2g, goff, frag, b1, b2, W3, b3, out);
}
